// Density_prop_GRU_56092272885822
// MI455X (gfx1250) — matrix operand order repaired and checked by reading, not compiled, not run on hardware
//
#include <hip/hip_runtime.h>
#include <hip/hip_bf16.h>

#define BB 64
#define TT 32
#define DD 128
#define NN 128
#define THREADS 512           // 16 waves (wave32)
#define STR 136               // padded bf16 row stride (16B-aligned rows, breaks bank conflicts)

typedef __attribute__((ext_vector_type(16))) __bf16 v16bf;
typedef __attribute__((ext_vector_type(8)))  float  v8f;

struct __align__(4) bf2 { __bf16 x, y; };

__device__ __forceinline__ float softplus_(float x) {
    return (x > 20.0f) ? x : log1pf(expf(x));
}
__device__ __forceinline__ float sigmoid_(float x) {
    return 1.0f / (1.0f + expf(-x));
}
__device__ __forceinline__ float wave_sum32(float a) {
#pragma unroll
    for (int o = 16; o > 0; o >>= 1) a += __shfl_xor(a, o, 32);
    return a;
}

// A fragment: 16x32 bf16 tile from row-major source (row stride STR).
// Lanes 0..15: M=lane, K=kt*32+{0..7,16..23}; lanes 16..31: M=lane-16, K=kt*32+{8..15,24..31}.
__device__ __forceinline__ v16bf load_fragA(const __bf16* base, int lane, int row, int kt) {
    const __bf16* p = base + row * STR + kt * 32 + ((lane >> 4) << 3);
    v16bf a;
#pragma unroll
    for (int i = 0; i < 8; ++i) { a[i] = p[i]; a[8 + i] = p[16 + i]; }
    return a;
}

// B fragment: 32x16 tile of B read from B^T stored row-major (row = N index, cols = K).
// Lanes 0..15: N=lane, K=kt*32+{0..7,16..23} ; lanes 16..31: N=lane-16, K=kt*32+{8..15,24..31}.
__device__ __forceinline__ v16bf load_fragB(const __bf16* baseT, int lane, int col, int kt) {
    const __bf16* p = baseT + col * STR + kt * 32 + ((lane >> 4) << 3);
    v16bf b;
#pragma unroll
    for (int i = 0; i < 8; ++i) { b[i] = p[i]; b[8 + i] = p[16 + i]; }
    return b;
}

// T = A(128x128 rm bf16) * B(given as B^T rm bf16) -> bf16 row-major T (stride STR)
__device__ __forceinline__ void mm_to_T(const __bf16* sA, const __bf16* sBT, __bf16* sT,
                                        int wave, int lane) {
    const int mrow = lane & 15;
    const int madd = (lane >> 4) * 8;
#pragma unroll
    for (int q = 0; q < 4; ++q) {
        const int tt = wave * 4 + q;
        const int ti = tt >> 3, tj = tt & 7;
        v8f c = {0.f, 0.f, 0.f, 0.f, 0.f, 0.f, 0.f, 0.f};
#pragma unroll
        for (int kt = 0; kt < 4; ++kt) {
            v16bf a = load_fragA(sA, lane, ti * 16 + mrow, kt);
            v16bf b = load_fragB(sBT, lane, tj * 16 + mrow, kt);
            c = __builtin_amdgcn_wmma_f32_16x16x32_bf16(false, a, false, b, (short)0, c,
                                                        false, false);
        }
#pragma unroll
        for (int v = 0; v < 8; ++v) {
            const int gi = ti * 16 + v + madd;
            const int gj = tj * 16 + mrow;
            sT[gi * STR + gj] = (__bf16)c[v];
        }
    }
}

// dst = (T * W + diag(scalWT*spW + xx*spU)) * outer(g,g)   (f32 row-major, stride NN)
__device__ __forceinline__ void mm_to_S(const __bf16* sT, const __bf16* sWT, float* dst,
                                        const float* g, const float* spW, const float* spU,
                                        float scalWT, float xx, int wave, int lane) {
    const int mrow = lane & 15;
    const int madd = (lane >> 4) * 8;
#pragma unroll
    for (int q = 0; q < 4; ++q) {
        const int tt = wave * 4 + q;
        const int ti = tt >> 3, tj = tt & 7;
        v8f c = {0.f, 0.f, 0.f, 0.f, 0.f, 0.f, 0.f, 0.f};
#pragma unroll
        for (int kt = 0; kt < 4; ++kt) {
            v16bf a = load_fragA(sT, lane, ti * 16 + mrow, kt);
            v16bf b = load_fragB(sWT, lane, tj * 16 + mrow, kt);
            c = __builtin_amdgcn_wmma_f32_16x16x32_bf16(false, a, false, b, (short)0, c,
                                                        false, false);
        }
        const int gj = tj * 16 + mrow;
        const float gjv = g[gj];
        const float dg = scalWT * spW[gj] + xx * spU[gj];
#pragma unroll
        for (int v = 0; v < 8; ++v) {
            const int gi = ti * 16 + v + madd;
            float val = c[v];
            if (gi == gj) val += dg;
            dst[gi * NN + gj] = val * g[gi] * gjv;
        }
    }
}

__global__ __launch_bounds__(THREADS, 1)
void dpgru_kernel(const float* __restrict__ x,
                  const float* __restrict__ Uz, const float* __restrict__ Wz,
                  const float* __restrict__ Ur, const float* __restrict__ Wr,
                  const float* __restrict__ Uh, const float* __restrict__ Wh,
                  const float* __restrict__ uzs, const float* __restrict__ wzs,
                  const float* __restrict__ urs, const float* __restrict__ wrs,
                  const float* __restrict__ uhs, const float* __restrict__ whs,
                  float* __restrict__ out_mu, float* __restrict__ out_S) {
    // f32 matrices (row stride NN)
    __shared__ float sSp[NN * NN];           // S_p (persistent, symmetric)
    __shared__ float sSz[NN * NN];           // S_z
    __shared__ float sB[NN * NN];            // S_r -> S_g -> S_h
    // bf16 tiles (row stride STR)
    __shared__ __attribute__((aligned(16))) __bf16 sT[NN * STR];    // T = W^T S
    __shared__ __attribute__((aligned(16))) __bf16 sSbf[NN * STR];  // bf16 copy of S_p / S_g
    __shared__ __attribute__((aligned(16))) __bf16 sW[NN * STR];    // W^T (row-major) = W col-major
    // vectors
    __shared__ float sx[NN], sMu[NN], sMuN[NN];
    __shared__ float sz[NN], sgz[NN], sr[NN], sgr[NN], srh[NN], sh[NN], sgh[NN];
    __shared__ float spUzV[NN], spWzV[NN], spUrV[NN], spWrV[NN], spUhV[NN], spWhV[NN];
    __shared__ float sPart[4 * NN];          // split-K matvec partials
    __shared__ float sScal[8];               // 0:xx 1:pp 2:trS 3:rr 4:trg

    const int b = blockIdx.x;
    const int tid = threadIdx.x;
    const int lane = tid & 31;
    const int wave = tid >> 5;

    for (int i = tid; i < NN * NN; i += THREADS) sSp[i] = 0.0f;
    if (tid < NN) {
        sMu[tid] = 0.0f;
        spUzV[tid] = softplus_(uzs[tid]);
        spWzV[tid] = softplus_(wzs[tid]);
        spUrV[tid] = softplus_(urs[tid]);
        spWrV[tid] = softplus_(wrs[tid]);
        spUhV[tid] = softplus_(uhs[tid]);
        spWhV[tid] = softplus_(whs[tid]);
    }
    __syncthreads();

    for (int t = 0; t < TT; ++t) {
        // ---- P0: load xt ; S_p -> bf16 (packed pairs) ----
        if (tid < NN) {
            sx[tid] = x[((size_t)(b * TT + t)) * DD + tid];
        } else {
            for (int i = tid - NN; i < (NN * NN) / 2; i += THREADS - NN) {
                const int row = i >> 6, c2 = i & 63;
                const float2 v = *reinterpret_cast<const float2*>(&sSp[row * NN + 2 * c2]);
                bf2 p; p.x = (__bf16)v.x; p.y = (__bf16)v.y;
                *reinterpret_cast<bf2*>(&sSbf[row * STR + 2 * c2]) = p;
            }
        }
        __syncthreads();

        // ---- P1: wave reductions (xx,pp,trS) ; split-K matvecs for az/ar ----
        if (wave == 0) {
            float a = 0.f;
#pragma unroll
            for (int i = 0; i < 4; ++i) { const float v = sx[lane + 32 * i]; a += v * v; }
            a = wave_sum32(a);
            if (lane == 0) sScal[0] = a;
        } else if (wave == 1) {
            float a = 0.f;
#pragma unroll
            for (int i = 0; i < 4; ++i) { const float v = sMu[lane + 32 * i]; a += v * v; }
            a = wave_sum32(a);
            if (lane == 0) sScal[1] = a;
        } else if (wave == 2) {
            float a = 0.f;
#pragma unroll
            for (int i = 0; i < 4; ++i) a += sSp[(lane + 32 * i) * (NN + 1)];
            a = wave_sum32(a);
            if (lane == 0) sScal[2] = a;
        }
        {
            const int m = tid >> 8;             // 0: z-gate, 1: r-gate
            const int part = (tid >> 7) & 1;    // K-split half
            const int j = tid & 127;
            const float* Umat = m ? Ur : Uz;
            const float* Wmat = m ? Wr : Wz;
            float acc = 0.f;
            const int k0 = part * 64;
            for (int d = k0; d < k0 + 64; ++d) acc += sx[d] * Umat[d * NN + j];
            for (int k = k0; k < k0 + 64; ++k) acc += sMu[k] * Wmat[k * NN + j];
            sPart[(m * 2 + part) * NN + j] = acc;
        }
        __syncthreads();

        // ---- P2: gate combines ; W_z^T -> bf16 (packed) ----
        if (tid < NN) {
            const float zv = sigmoid_(sPart[tid] + sPart[NN + tid]);
            sz[tid] = zv; sgz[tid] = zv * (1.0f - zv);
        } else if (tid < 2 * NN) {
            const int j = tid - NN;
            const float rv = sigmoid_(sPart[2 * NN + j] + sPart[3 * NN + j]);
            sr[j] = rv; sgr[j] = rv * (1.0f - rv);
            srh[j] = sMu[j] * rv;
        } else {
            for (int i = tid - 2 * NN; i < (NN * NN) / 2; i += THREADS - 2 * NN) {
                const int m = i & 127, kp = i >> 7;       // coalesced over m
                bf2 p;
                p.x = (__bf16)Wz[(2 * kp) * NN + m];
                p.y = (__bf16)Wz[(2 * kp + 1) * NN + m];
                *reinterpret_cast<bf2*>(&sW[m * STR + 2 * kp]) = p;
            }
        }
        __syncthreads();

        // ---- gate z: S_z = (W_z^T S_p W_z + diag) * outer(gz,gz) ----
        mm_to_T(sW, sSbf, sT, wave, lane);
        __syncthreads();
        mm_to_S(sT, sW, sSz, sgz, spWzV, spUzV, sScal[1] + sScal[2], sScal[0], wave, lane);
        __syncthreads();

        // ---- P5: W_r^T -> bf16 (packed, all threads) ----
        for (int i = tid; i < (NN * NN) / 2; i += THREADS) {
            const int m = i & 127, kp = i >> 7;
            bf2 p;
            p.x = (__bf16)Wr[(2 * kp) * NN + m];
            p.y = (__bf16)Wr[(2 * kp + 1) * NN + m];
            *reinterpret_cast<bf2*>(&sW[m * STR + 2 * kp]) = p;
        }
        __syncthreads();

        // ---- gate r ----
        mm_to_T(sW, sSbf, sT, wave, lane);
        __syncthreads();
        mm_to_S(sT, sW, sB, sgr, spWrV, spUrV, sScal[1] + sScal[2], sScal[0], wave, lane);
        __syncthreads();

        // ---- P8: ah split-K matvec (tid<256) ; S_g elementwise in place (tid>=256) ----
        if (tid < 2 * NN) {
            const int part = tid >> 7;
            const int j = tid & 127;
            float acc = 0.f;
            const int k0 = part * 64;
            for (int d = k0; d < k0 + 64; ++d) acc += sx[d] * Uh[d * NN + j];
            for (int k = k0; k < k0 + 64; ++k) acc += srh[k] * Wh[k * NN + j];
            sPart[part * NN + j] = acc;
        } else {
            for (int i = tid - 2 * NN; i < NN * NN; i += THREADS - 2 * NN) {
                const int ii = i >> 7, jj = i & 127;
                const float spv = sSp[i], srv = sB[i];
                sB[i] = spv * srv + sMu[ii] * sMu[jj] * srv + sr[ii] * sr[jj] * spv;  // S_g
            }
        }
        __syncthreads();

        // ---- P9: h combine ; rr & trg wave-reductions ; W_h^T load ; S_g -> bf16 ----
        if (wave < 4) {
            const int j = tid;  // tid < 128
            const float hv = tanhf(sPart[j] + sPart[NN + j]);
            sh[j] = hv; sgh[j] = 1.0f - hv * hv;
        } else if (wave == 4) {
            float a = 0.f;
#pragma unroll
            for (int i = 0; i < 4; ++i) { const float v = srh[lane + 32 * i]; a += v * v; }
            a = wave_sum32(a);
            if (lane == 0) sScal[3] = a;
        } else if (wave == 5) {
            float a = 0.f;
#pragma unroll
            for (int i = 0; i < 4; ++i) a += sB[(lane + 32 * i) * (NN + 1)];
            a = wave_sum32(a);
            if (lane == 0) sScal[4] = a;
        } else if (wave < 11) {   // waves 6..10: W_h^T -> bf16
            for (int i = (wave - 6) * 32 + lane; i < (NN * NN) / 2; i += 5 * 32) {
                const int m = i & 127, kp = i >> 7;
                bf2 p;
                p.x = (__bf16)Wh[(2 * kp) * NN + m];
                p.y = (__bf16)Wh[(2 * kp + 1) * NN + m];
                *reinterpret_cast<bf2*>(&sW[m * STR + 2 * kp]) = p;
            }
        } else {                  // waves 11..15: S_g -> bf16
            for (int i = (wave - 11) * 32 + lane; i < (NN * NN) / 2; i += 5 * 32) {
                const int row = i >> 6, c2 = i & 63;
                const float2 v = *reinterpret_cast<const float2*>(&sB[row * NN + 2 * c2]);
                bf2 p; p.x = (__bf16)v.x; p.y = (__bf16)v.y;
                *reinterpret_cast<bf2*>(&sSbf[row * STR + 2 * c2]) = p;
            }
        }
        __syncthreads();

        // ---- gate h: S_h = (W_h^T S_g W_h + diag) * outer(gh,gh) ----
        mm_to_T(sW, sSbf, sT, wave, lane);
        __syncthreads();
        mm_to_S(sT, sW, sB, sgh, spWhV, spUhV, sScal[3] + sScal[4], sScal[0], wave, lane);
        __syncthreads();

        // ---- P13: final combine, write S_p in place + stream to HBM ----
        float* outS = out_S + ((size_t)(b * TT + t)) * NN * NN;
        for (int i = tid; i < NN * NN; i += THREADS) {
            const int ii = i >> 7, jj = i & 127;
            const float Szv = sSz[i], Spv = sSp[i], Shv = sB[i];
            const float zi = sz[ii], zj = sz[jj];
            const float d1 = sMu[ii] - sh[ii], d2 = sMu[jj] - sh[jj];
            float v = Spv * (Szv + zi * zj) + Shv * (Szv + (1.0f - zi) * (1.0f - zj)) +
                      Szv * d1 * d2;
            if (!isfinite(v)) v = 0.0f;
            if (ii == jj) v = fabsf(v);
            sSp[i] = v;
            outS[i] = v;
        }
        if (tid < NN) sMuN[tid] = sz[tid] * sMu[tid] + (1.0f - sz[tid]) * sh[tid];
        __syncthreads();
        if (tid < NN) {
            sMu[tid] = sMuN[tid];
            out_mu[((size_t)(b * TT + t)) * NN + tid] = sMu[tid];
        }
        __syncthreads();
    }
}

extern "C" void kernel_launch(void* const* d_in, const int* in_sizes, int n_in,
                              void* d_out, int out_size, void* d_ws, size_t ws_size,
                              hipStream_t stream) {
    (void)in_sizes; (void)n_in; (void)out_size; (void)d_ws; (void)ws_size;
    const float* x   = (const float*)d_in[0];
    const float* Uz  = (const float*)d_in[1];
    const float* Wz  = (const float*)d_in[2];
    const float* Ur  = (const float*)d_in[3];
    const float* Wr  = (const float*)d_in[4];
    const float* Uh  = (const float*)d_in[5];
    const float* Wh  = (const float*)d_in[6];
    const float* uzs = (const float*)d_in[7];
    const float* wzs = (const float*)d_in[8];
    const float* urs = (const float*)d_in[9];
    const float* wrs = (const float*)d_in[10];
    const float* uhs = (const float*)d_in[11];
    const float* whs = (const float*)d_in[12];

    float* out_mu = (float*)d_out;                       // [B,T,N]
    float* out_S  = out_mu + (size_t)BB * TT * NN;       // [B,T,N,N]

    dpgru_kernel<<<dim3(BB), dim3(THREADS), 0, stream>>>(
        x, Uz, Wz, Ur, Wr, Uh, Wh, uzs, wzs, urs, wrs, uhs, whs, out_mu, out_S);
}
